// PicoDetHead_30528627540208
// MI455X (gfx1250) — matrix operand order repaired and checked by reading, not compiled, not run on hardware
//
#include <hip/hip_runtime.h>
#include <hip/hip_bf16.h>

typedef __attribute__((ext_vector_type(16))) __bf16 v16bf;
typedef __attribute__((ext_vector_type(8)))  __bf16 bhalf8;
typedef __attribute__((ext_vector_type(8)))  float  v8f;
typedef unsigned long long u64;

#define CSTR 104   // bf16 channel stride for 96-ch LDS buffers / weights (16B aligned, conflict-free)
#define YSTR 56    // bf16 channel stride for 48-ch y buffer

// LDS partition (bytes), total < 320KB per WGP
#define OFF_FEAT 0         // 400*104 bf16  (20x20 halo-2 feat tile)
#define OFF_CAT  83200     // 336*104 bf16  (18x18 cat1, padded to 21 N-tiles)
#define OFF_CAT2 153088    // 256*104 bf16  (16x16 cat2)
#define OFF_Y    206336    // 400*56  bf16  (ghost primary out)
#define OFF_W    251136    // up to 96*104 bf16 weights slab
#define OFF_FP   271104    // 1248 floats (affine + dw weights / head biases)
#define OFF_LTRB 276096    // 256*4 floats
#define OFF_PIX  280192    // 256 u64 (cls max keys)
#define OFF_OBJ  282240    // 256 floats
#define OFF_M0   283264    // 400 bytes valid mask R0
#define OFF_M1   283664    // 336 bytes valid mask R1
#define OFF_M2   284000    // 256 bytes valid mask R2
#define SMEM_BYTES 284288

#define NPIX 8400
#define SCORE_TH 0.05f
#define IOU_TH   0.6f

struct HeadArgs {
  const float* feat;
  const float* pw[2][2]; const float* ps[2][2]; const float* pb[2][2];
  const float* cw[2][2]; const float* cs[2][2]; const float* cb[2][2];
  const float *clsw, *clsb, *objw, *objb, *regw, *regb;
  float* boxes; float* conf; int* label;
  int H, W, tilesX, levelBase; float stride;
};

__device__ __forceinline__ unsigned fkey(float f) {
  unsigned u = __float_as_uint(f);
  return (u & 0x80000000u) ? ~u : (u | 0x80000000u);
}
__device__ __forceinline__ float unfkey(unsigned k) {
  unsigned u = (k & 0x80000000u) ? (k ^ 0x80000000u) : ~k;
  return __uint_as_float(u);
}

// A fragment: lane holds row m, K in {kh*8..+8} U {16+kh*8..+8} (16-bit A layout)
__device__ __forceinline__ v16bf load_afrag(const __bf16* row, int kk, int kh) {
  const __bf16* p = row + kk*32 + kh*8;
  bhalf8 vlo = *(const bhalf8*)p;
  bhalf8 vhi = *(const bhalf8*)(p + 16);
  return __builtin_shufflevector(vlo, vhi, 0,1,2,3,4,5,6,7,8,9,10,11,12,13,14,15);
}
// B fragment: lane holds col n, K in {kh*8..+8} U {16+kh*8..+8} (matches A layout)
__device__ __forceinline__ v16bf load_bfrag(const __bf16* row, int kk, int kh) {
  const __bf16* p = row + kk*32 + kh*8;
  bhalf8 vlo = *(const bhalf8*)p;
  bhalf8 vhi = *(const bhalf8*)(p + 16);
  return __builtin_shufflevector(vlo, vhi, 0,1,2,3,4,5,6,7,8,9,10,11,12,13,14,15);
}

// Ghost primary: y[48][pix] = clip((W(48x96) @ x) * ps + pb, 0, 6), zero outside image
__device__ void gemm_ghost(const __bf16* act, const __bf16* w,
                           const float* ps, const float* pb,
                           __bf16* yout, const unsigned char* vmask,
                           int NT, int wave, int lane)
{
  int kh = lane >> 4, ln = lane & 15;
  for (int t = wave; t < 3*NT; t += 8) {
    int mt = t / NT, nt = t - mt*NT;
    v8f acc = {0.f,0.f,0.f,0.f,0.f,0.f,0.f,0.f};
#pragma unroll
    for (int kk = 0; kk < 3; ++kk) {
      v16bf a  = load_afrag(w   + (mt*16 + ln)*CSTR, kk, kh);
      v16bf bm = load_bfrag(act + (nt*16 + ln)*CSTR, kk, kh);
      acc = __builtin_amdgcn_wmma_f32_16x16x32_bf16(false, a, false, bm, (short)0, acc, false, false);
    }
    int pix = nt*16 + ln;
    int mb  = mt*16 + kh*8;
    bool v = vmask[pix] != 0;
    bhalf8 o;
#pragma unroll
    for (int i = 0; i < 8; ++i) {
      float x = acc[i]*ps[mb+i] + pb[mb+i];
      x = fminf(fmaxf(x, 0.f), 6.f);
      o[i] = v ? (__bf16)x : (__bf16)0.f;
    }
    *(bhalf8*)&yout[pix*YSTR + mb] = o;   // aligned ds_store_b128
  }
}

// Ghost cheap: copy y into cat[0..48) and depthwise 3x3 -> cat[48..96)
template<int WI>
__device__ void ghost_cheap(const __bf16* yb, const float* cw, const float* cs, const float* cb,
                            __bf16* cat, const unsigned char* vout, int tid)
{
  const int WO = WI - 2, NO = WO*WO;
  for (int idx = tid; idx < NO*48; idx += 256) {
    int p = idx / 48, c = idx - p*48;
    int oy = p / WO, ox = p - oy*WO;
    cat[p*CSTR + c] = yb[((oy+1)*WI + (ox+1))*YSTR + c];
    float s = 0.f;
#pragma unroll
    for (int ky = 0; ky < 3; ++ky)
#pragma unroll
      for (int kx = 0; kx < 3; ++kx)
        s += cw[c*9 + ky*3 + kx] * (float)yb[((oy+ky)*WI + (ox+kx))*YSTR + c];
    s = s * cs[c] + cb[c];
    s = fminf(fmaxf(s, 0.f), 6.f);
    cat[p*CSTR + 48 + c] = vout[p] ? (__bf16)s : (__bf16)0.f;
  }
}

// cls head (80x96) with fused per-pixel max/argmax via packed u64 LDS atomic-max
__device__ void gemm_cls(const __bf16* cat2, const __bf16* w, const float* clsb,
                         u64* pixmax, int wave, int lane)
{
  int kh = lane >> 4, ln = lane & 15;
  for (int t = wave; t < 80; t += 8) {          // 5 M-tiles x 16 N-tiles
    int mt = t >> 4, nt = t & 15;
    v8f acc = {0.f,0.f,0.f,0.f,0.f,0.f,0.f,0.f};
#pragma unroll
    for (int kk = 0; kk < 3; ++kk) {
      v16bf a  = load_afrag(w    + (mt*16 + ln)*CSTR, kk, kh);
      v16bf bm = load_bfrag(cat2 + (nt*16 + ln)*CSTR, kk, kh);
      acc = __builtin_amdgcn_wmma_f32_16x16x32_bf16(false, a, false, bm, (short)0, acc, false, false);
    }
    int pix = nt*16 + ln;
    int mb  = mt*16 + kh*8;
    float best = -3.0e38f; int bc = 0;
#pragma unroll
    for (int i = 0; i < 8; ++i) {
      float l = acc[i] + clsb[mb+i];
      if (l > best) { best = l; bc = mb+i; }
    }
    u64 key = ((u64)fkey(best) << 32) | (u64)(255 - bc);  // smaller class wins ties
    atomicMax(&pixmax[pix], key);
  }
}

// reg head (32x96) with fused DFL: each lane's 8 accs == one side's 8 bins
__device__ void gemm_reg(const __bf16* cat2, const __bf16* w, const float* regb,
                         float* ltrb, float stride, int wave, int lane)
{
  int kh = lane >> 4, ln = lane & 15;
  for (int t = wave; t < 32; t += 8) {          // 2 M-tiles x 16 N-tiles
    int mt = t >> 4, nt = t & 15;
    v8f acc = {0.f,0.f,0.f,0.f,0.f,0.f,0.f,0.f};
#pragma unroll
    for (int kk = 0; kk < 3; ++kk) {
      v16bf a  = load_afrag(w    + (mt*16 + ln)*CSTR, kk, kh);
      v16bf bm = load_bfrag(cat2 + (nt*16 + ln)*CSTR, kk, kh);
      acc = __builtin_amdgcn_wmma_f32_16x16x32_bf16(false, a, false, bm, (short)0, acc, false, false);
    }
    int pix  = nt*16 + ln;
    int side = mt*2 + kh;
    float l[8]; float mx = -3.0e38f;
#pragma unroll
    for (int i = 0; i < 8; ++i) { l[i] = acc[i] + regb[side*8+i]; mx = fmaxf(mx, l[i]); }
    float den = 0.f, num = 0.f;
#pragma unroll
    for (int i = 0; i < 8; ++i) { float e = expf(l[i]-mx); den += e; num += e*(float)i; }
    ltrb[pix*4 + side] = (num/den) * stride;
  }
}

__device__ __forceinline__ void load_w96(const float* g, __bf16* l, int O, int tid) {
  for (int i = tid; i < O*96; i += 256) { int o = i/96, c = i - o*96; l[o*CSTR + c] = (__bf16)g[i]; }
}

__global__ __launch_bounds__(256, 1)
void head_kernel(HeadArgs A)
{
  extern __shared__ char smem[];
  __bf16* featL = (__bf16*)(smem + OFF_FEAT);
  __bf16* catL  = (__bf16*)(smem + OFF_CAT);
  __bf16* cat2L = (__bf16*)(smem + OFF_CAT2);
  __bf16* ybuf  = (__bf16*)(smem + OFF_Y);
  __bf16* wslab = (__bf16*)(smem + OFF_W);
  float*  fslab = (float*)(smem + OFF_FP);
  float*  ltrb  = (float*)(smem + OFF_LTRB);
  u64*    pixmax= (u64*)(smem + OFF_PIX);
  float*  objv  = (float*)(smem + OFF_OBJ);
  unsigned char* m0 = (unsigned char*)(smem + OFF_M0);
  unsigned char* m1 = (unsigned char*)(smem + OFF_M1);
  unsigned char* m2 = (unsigned char*)(smem + OFF_M2);

  const int tid = threadIdx.x, wave = tid >> 5, lane = tid & 31;
  const int b = blockIdx.y;
  const int tx = blockIdx.x % A.tilesX, ty = blockIdx.x / A.tilesX;
  const int x0 = tx*16, y0 = ty*16;
  const int H = A.H, W = A.W;

  // validity masks for the three regions (R0=20x20 halo2, R1=18x18, R2=16x16)
  for (int i = tid; i < 400; i += 256) {
    int py = i/20, px = i - py*20; int gx = x0-2+px, gy = y0-2+py;
    m0[i] = (gx >= 0 && gy >= 0 && gx < W && gy < H) ? 1 : 0;
  }
  for (int i = tid; i < 336; i += 256) {
    unsigned char v = 0;
    if (i < 324) { int py = i/18, px = i - py*18; int gx = x0-1+px, gy = y0-1+py;
                   v = (gx >= 0 && gy >= 0 && gx < W && gy < H) ? 1 : 0; }
    m1[i] = v;
  }
  { int py = tid >> 4, px = tid & 15; m2[tid] = (x0+px < W && y0+py < H) ? 1 : 0; }

  // load feat tile (fp32 -> bf16), zero-pad outside image; pixel-major LDS layout
  const float* fb = A.feat + (size_t)b * 96 * H * W;
  for (int i = tid; i < 96*400; i += 256) {
    int c = i / 400, p = i - c*400;
    int py = p / 20, px = p - py*20;
    int gx = x0-2+px, gy = y0-2+py;
    float v = (gx >= 0 && gy >= 0 && gx < W && gy < H)
            ? fb[(size_t)c*H*W + (size_t)gy*W + gx] : 0.f;
    featL[p*CSTR + c] = (__bf16)v;
  }
  __syncthreads();

  for (int tower = 0; tower < 2; ++tower) {
    // stage tower weights (bf16) + affine/dw params (fp32) into LDS
    load_w96(A.pw[tower][0], wslab, 48, tid);
    load_w96(A.pw[tower][1], wslab + 48*CSTR, 48, tid);
    for (int k = 0; k < 2; ++k) {
      float* f = fslab + k*624;
      const float *ps = A.ps[tower][k], *pb = A.pb[tower][k];
      const float *cs = A.cs[tower][k], *cb = A.cb[tower][k], *cw = A.cw[tower][k];
      for (int i = tid; i < 48; i += 256) { f[i]=ps[i]; f[48+i]=pb[i]; f[96+i]=cs[i]; f[144+i]=cb[i]; }
      for (int i = tid; i < 432; i += 256) f[192+i] = cw[i];
    }
    __syncthreads();

    gemm_ghost(featL, wslab, fslab, fslab+48, ybuf, m0, 25, wave, lane);     // block0 primary on R0
    __syncthreads();
    ghost_cheap<20>(ybuf, fslab+192, fslab+96, fslab+144, catL, m1, tid);    // block0 cheap -> cat1 (R1)
    __syncthreads();
    gemm_ghost(catL, wslab + 48*CSTR, fslab+624, fslab+624+48, ybuf, m1, 21, wave, lane); // block1 primary
    __syncthreads();
    ghost_cheap<18>(ybuf, fslab+624+192, fslab+624+96, fslab+624+144, cat2L, m2, tid);    // -> cat2 (R2)
    __syncthreads();

    if (tower == 0) {
      pixmax[tid] = 0ull;
      load_w96(A.clsw, wslab, 80, tid);
      for (int i = tid; i < 80; i += 256) fslab[i]       = A.clsb[i];
      for (int i = tid; i < 96; i += 256) fslab[128 + i] = A.objw[i];
      if (tid == 0) fslab[224] = A.objb[0];
      __syncthreads();
      gemm_cls(cat2L, wslab, fslab, pixmax, wave, lane);
      { // obj head: one pixel per thread
        float s = fslab[224];
        for (int c = 0; c < 96; ++c) s += fslab[128 + c] * (float)cat2L[tid*CSTR + c];
        objv[tid] = s;
      }
      __syncthreads();
    } else {
      load_w96(A.regw, wslab, 32, tid);
      for (int i = tid; i < 32; i += 256) fslab[i] = A.regb[i];
      __syncthreads();
      gemm_reg(cat2L, wslab, fslab, ltrb, A.stride, wave, lane);
      __syncthreads();
    }
  }

  // decode & write per-pixel outputs (tid == pixel)
  {
    int px = tid & 15, py = tid >> 4;
    int gx = x0 + px, gy = y0 + py;
    if (gx < W && gy < H) {
      u64 k = pixmax[tid];
      float clsl = unfkey((unsigned)(k >> 32));
      int lab = 255 - (int)(k & 0xFFull);
      float cf = 1.f / (1.f + expf(-(clsl + objv[tid])));
      if (cf < SCORE_TH) cf = 0.f;
      int n = A.levelBase + gy*W + gx;
      size_t o = (size_t)b * NPIX + n;
      float ax = (gx + 0.5f) * A.stride, ay = (gy + 0.5f) * A.stride;
      A.boxes[o*4+0] = ax - ltrb[tid*4+0];
      A.boxes[o*4+1] = ay - ltrb[tid*4+1];
      A.boxes[o*4+2] = ax + ltrb[tid*4+2];
      A.boxes[o*4+3] = ay + ltrb[tid*4+3];
      A.conf[o]  = cf;
      A.label[o] = lab;
    }
  }
}

// class-agnostic sequential NMS, one workgroup per image
__global__ __launch_bounds__(256, 1)
void nms_kernel(const float* boxes, float* conf, const int* label, float* out, int B)
{
  __shared__ u64 red[256];
  __shared__ float sb[5];
  __shared__ int sidx;
  const int b = blockIdx.x, tid = threadIdx.x;
  const float* bx = boxes + (size_t)b * NPIX * 4;
  float* cf = conf + (size_t)b * NPIX;
  const size_t scOff = (size_t)B * 400, lbOff = (size_t)B * 500;

  for (int d = 0; d < 100; ++d) {
    u64 best = 0;
    for (int i = tid; i < NPIX; i += 256) {
      u64 k = ((u64)fkey(cf[i]) << 32) | (u64)(16383 - i);  // smaller idx wins ties
      if (k > best) best = k;
    }
    red[tid] = best; __syncthreads();
    for (int s = 128; s > 0; s >>= 1) {
      if (tid < s && red[tid+s] > red[tid]) red[tid] = red[tid+s];
      __syncthreads();
    }
    if (tid == 0) {
      int i = 16383 - (int)(red[0] & 0x3FFFull);
      sidx = i; sb[4] = cf[i];
      sb[0] = bx[i*4+0]; sb[1] = bx[i*4+1]; sb[2] = bx[i*4+2]; sb[3] = bx[i*4+3];
    }
    __syncthreads();
    bool valid = sb[4] >= SCORE_TH;
    if (tid == 0) {
      size_t od = (size_t)b * 100 + d;
      if (valid) {
        out[od*4+0]=sb[0]; out[od*4+1]=sb[1]; out[od*4+2]=sb[2]; out[od*4+3]=sb[3];
        out[scOff+od] = sb[4];
        out[lbOff+od] = (float)label[(size_t)b*NPIX + sidx];
      } else {
        out[od*4+0]=0.f; out[od*4+1]=0.f; out[od*4+2]=0.f; out[od*4+3]=0.f;
        out[scOff+od] = 0.f;
        out[lbOff+od] = -1.f;
      }
    }
    if (valid) {
      float ax1=sb[0], ay1=sb[1], ax2=sb[2], ay2=sb[3];
      float areaA = (ax2-ax1)*(ay2-ay1);
      for (int i = tid; i < NPIX; i += 256) {
        float x1=bx[i*4+0], y1=bx[i*4+1], x2=bx[i*4+2], y2=bx[i*4+3];
        float ix1=fmaxf(ax1,x1), iy1=fmaxf(ay1,y1), ix2=fminf(ax2,x2), iy2=fminf(ay2,y2);
        float inter = fmaxf(ix2-ix1, 0.f) * fmaxf(iy2-iy1, 0.f);
        float areaB = (x2-x1)*(y2-y1);
        float iou = inter / (areaA + areaB - inter + 1e-7f);
        if (iou > IOU_TH) cf[i] = 0.f;
      }
      if (tid == 0) cf[sidx] = 0.f;
    }
    __syncthreads();
  }
}

extern "C" void kernel_launch(void* const* d_in, const int* in_sizes, int n_in,
                              void* d_out, int out_size, void* d_ws, size_t ws_size,
                              hipStream_t stream)
{
  (void)n_in; (void)out_size; (void)ws_size;
  const int B = in_sizes[0] / (96 * 80 * 80);

  float* boxes = (float*)d_ws;
  float* conf  = boxes + (size_t)B * NPIX * 4;
  int*   label = (int*)(conf + (size_t)B * NPIX);

  const int   Hs[3]     = {80, 40, 20};
  const float strides[3]= {8.f, 16.f, 32.f};
  const int   bases[3]  = {0, 6400, 8000};

  for (int L = 0; L < 3; ++L) {
    HeadArgs A;
    A.feat = (const float*)d_in[L];
    for (int k = 0; k < 2; ++k) {
      int cb0 = 9 + 6*k, rb0 = 33 + 6*k;        // cls_tower / reg_tower: cb,cs,cw,pb,ps,pw
      A.cb[0][k]=(const float*)d_in[cb0+0]; A.cs[0][k]=(const float*)d_in[cb0+1];
      A.cw[0][k]=(const float*)d_in[cb0+2]; A.pb[0][k]=(const float*)d_in[cb0+3];
      A.ps[0][k]=(const float*)d_in[cb0+4]; A.pw[0][k]=(const float*)d_in[cb0+5];
      A.cb[1][k]=(const float*)d_in[rb0+0]; A.cs[1][k]=(const float*)d_in[rb0+1];
      A.cw[1][k]=(const float*)d_in[rb0+2]; A.pb[1][k]=(const float*)d_in[rb0+3];
      A.ps[1][k]=(const float*)d_in[rb0+4]; A.pw[1][k]=(const float*)d_in[rb0+5];
    }
    A.clsb = (const float*)d_in[3 + 2*L];  A.clsw = (const float*)d_in[4 + 2*L];
    A.objb = (const float*)d_in[21 + 2*L]; A.objw = (const float*)d_in[22 + 2*L];
    A.regb = (const float*)d_in[27 + 2*L]; A.regw = (const float*)d_in[28 + 2*L];
    A.boxes = boxes; A.conf = conf; A.label = label;
    A.H = Hs[L]; A.W = Hs[L];
    A.tilesX = (Hs[L] + 15) / 16;
    A.levelBase = bases[L];
    A.stride = strides[L];
    int tiles = A.tilesX * A.tilesX;
    head_kernel<<<dim3(tiles, B), dim3(256), SMEM_BYTES, stream>>>(A);
  }
  nms_kernel<<<dim3(B), dim3(256), 0, stream>>>(boxes, conf, label, (float*)d_out, B);
}
